// DecoderAttention_86560770883869
// MI455X (gfx1250) — matrix operand order repaired and checked by reading, not compiled, not run on hardware
//
#include <hip/hip_runtime.h>
#include <hip/hip_bf16.h>

// ---------------------------------------------------------------------------
// Decoder attention, fused flash-attention style, bf16 WMMA (gfx1250 CDNA5).
//   q = enc @ Qs + Qb (x 1/8), k = enc @ Ks + Kb, v = resid @ Vs + Vb
//   attn = softmax(causal_mask(q k^T))   -- flash/online softmax in registers
//   z = attn @ v ; out = z @ O + Ob
// All GEMMs: v_wmma_f32_16x16x32_bf16. K-depth 64 per LDS stage, packed-b32
// LDS transposes, async global->LDS staging (ASYNCcnt path) when available.
// ---------------------------------------------------------------------------

typedef __bf16 bf16t;
typedef __attribute__((ext_vector_type(16))) __bf16 v16bf;
typedef __attribute__((ext_vector_type(8)))  __bf16 v8bf;
typedef __attribute__((ext_vector_type(4)))  __bf16 v4bf;
typedef __attribute__((ext_vector_type(8)))  float  v8f;
typedef __attribute__((ext_vector_type(4)))  int    v4i;

#define N_HEADS 16
#define D_MODEL 1024
#define D_HEAD  64
#define BATCH   8
#define SEQ     1024
#define BS      (BATCH*SEQ)
#define IGNORE_VAL (-100000.0f)

#if defined(__AMDGCN__) && __has_builtin(__builtin_amdgcn_global_load_async_to_lds_b128)
#define ASYNC_STAGE 1
#else
#define ASYNC_STAGE 0
#endif

// ---------------- helpers --------------------------------------------------

__device__ __forceinline__ v8f zero_v8f() {
  v8f z;
#pragma unroll
  for (int i = 0; i < 8; ++i) z[i] = 0.0f;
  return z;
}

__device__ __forceinline__ v8f wmma_bf16(v16bf a, v16bf b, v8f c) {
  return __builtin_amdgcn_wmma_f32_16x16x32_bf16(
      false, a, false, b, (short)0, c, false, false);
}

// A-fragment (16x32 MxK): lane&15 = row; lanes<16 hold K {0..7,16..23},
// lanes>=16 hold K {8..15,24..31}. rowptr -> 32 contiguous K elements.
__device__ __forceinline__ v16bf load_afrag(const bf16t* rowptr, int half) {
  union { v16bf v; v8bf s[2]; } u;
  u.s[0] = *(const v8bf*)(rowptr + half * 8);
  u.s[1] = *(const v8bf*)(rowptr + 16 + half * 8);
  return u.v;
}

// B-fragment (32x16 KxN): lane&15 = column N; lanes<16 K {0..7,16..23},
// lanes>=16 K {8..15,24..31}. nrowptr -> 32 contiguous K elements for column N.
__device__ __forceinline__ v16bf load_bfrag(const bf16t* nrowptr, int half) {
  union { v16bf v; v8bf s[2]; } u;
  u.s[0] = *(const v8bf*)(nrowptr + half * 8);
  u.s[1] = *(const v8bf*)(nrowptr + 16 + half * 8);
  return u.v;
}

__device__ __forceinline__ unsigned pack2(bf16t lo, bf16t hi) {
  union { unsigned u; bf16t b[2]; } p;
  p.b[0] = lo; p.b[1] = hi;
  return p.u;
}

// 16B global -> LDS stage; async path uses GLOBAL_LOAD_ASYNC_TO_LDS_B128.
// (Builtin signature per hipcc diagnostic: (AS1 v4i* src, AS3 v4i* dst,
//  imm offset, imm cpol).)
__device__ __forceinline__ void stage16(bf16t* lds_dst, const bf16t* gsrc) {
#if ASYNC_STAGE
  __builtin_amdgcn_global_load_async_to_lds_b128(
      (__attribute__((address_space(1))) v4i*)gsrc,
      (__attribute__((address_space(3))) v4i*)lds_dst, 0, 0);
#else
  *(v8bf*)lds_dst = *(const v8bf*)gsrc;
#endif
}

__device__ __forceinline__ void wait_stage() {
#if ASYNC_STAGE
  asm volatile("s_wait_asynccnt 0x0" ::: "memory");
#endif
}

// ---------------- f32 -> bf16 conversion ----------------------------------

__global__ void cvt_f32_bf16_v4(const float4* __restrict__ src,
                                v4bf* __restrict__ dst, int n4) {
  int i = blockIdx.x * blockDim.x + threadIdx.x;
  if (i < n4) {
    float4 f = src[i];
    v4bf o;
    o[0] = (bf16t)f.x; o[1] = (bf16t)f.y; o[2] = (bf16t)f.z; o[3] = (bf16t)f.w;
    dst[i] = o;
  }
}

// ---------------- QKV projection ------------------------------------------
// grid: (BS/64, H, 3)  block: 256. 64x64 output tile, K staged 64 deep.
__global__ __launch_bounds__(256) void proj_qkv(
    const bf16t* __restrict__ enc, const bf16t* __restrict__ resid,
    const bf16t* __restrict__ Wq, const bf16t* __restrict__ Wk,
    const bf16t* __restrict__ Wv,
    const float* __restrict__ Qb, const float* __restrict__ Kb,
    const float* __restrict__ Vb,
    bf16t* __restrict__ qo, bf16t* __restrict__ ko, bf16t* __restrict__ vo) {
  __shared__ __align__(64) bf16t Als[64][64];   // activations [M][K]
  __shared__ __align__(64) bf16t Wt[64][64];    // weights transposed [N][K]

  const int t    = threadIdx.x;
  const int wv   = t >> 5;
  const int lane = t & 31;
  const int half = lane >> 4;
  const int lr   = lane & 15;
  const int h    = blockIdx.y;
  const int type = blockIdx.z;

  const bf16t* X    = (type == 2) ? resid : enc;
  const bf16t* W    = (type == 0) ? Wq : ((type == 1) ? Wk : Wv);
  const float* bias = (type == 0) ? Qb : ((type == 1) ? Kb : Vb);
  bf16t*       P    = (type == 0) ? qo : ((type == 1) ? ko : vo);
  const float  osc  = (type == 0) ? 0.125f : 1.0f;   // 1/sqrt(64) into q

  const int m0  = blockIdx.x * 64;
  const int mi  = (wv * 2) >> 2;
  const int ni0 = (wv * 2) & 3;
  const int ni1 = ni0 + 1;

  v8f c0 = zero_v8f(), c1 = zero_v8f();
  const size_t wbase = (size_t)h * D_MODEL * D_HEAD;

  for (int k0 = 0; k0 < D_MODEL; k0 += 64) {
    __syncthreads();
    { // stage A: 64 rows x 64 K, 32B per thread (async when available)
      int row = t >> 2, c = (t & 3) * 16;
      const bf16t* src = &X[(size_t)(m0 + row) * D_MODEL + k0 + c];
      stage16(&Als[row][c], src);
      stage16(&Als[row][c + 8], src + 8);
    }
    { // stage W transposed, packed b32 stores: rows 2p,2p+1 -> Wt[n][2p..2p+1]
      int p = t >> 3, ch = (t & 7) * 8;
      const bf16t* wp = &W[wbase + (size_t)(k0 + 2 * p) * D_HEAD + ch];
      v8bf r0 = *(const v8bf*)wp;
      v8bf r1 = *(const v8bf*)(wp + D_HEAD);
#pragma unroll
      for (int j = 0; j < 8; ++j)
        *(unsigned*)&Wt[ch + j][2 * p] = pack2(r0[j], r1[j]);
    }
    wait_stage();
    __syncthreads();

    v16bf a0 = load_afrag(&Als[mi * 16 + lr][0],  half);
    v16bf a1 = load_afrag(&Als[mi * 16 + lr][32], half);
    c0 = wmma_bf16(a0, load_bfrag(&Wt[ni0 * 16 + lr][0],  half), c0);
    c0 = wmma_bf16(a1, load_bfrag(&Wt[ni0 * 16 + lr][32], half), c0);
    c1 = wmma_bf16(a0, load_bfrag(&Wt[ni1 * 16 + lr][0],  half), c1);
    c1 = wmma_bf16(a1, load_bfrag(&Wt[ni1 * 16 + lr][32], half), c1);
  }

  const float bias0 = bias[h * D_HEAD + ni0 * 16 + lr];
  const float bias1 = bias[h * D_HEAD + ni1 * 16 + lr];
#pragma unroll
  for (int r = 0; r < 8; ++r) {
    int g  = m0 + mi * 16 + r + 8 * half;
    int b_ = g >> 10, s_ = g & (SEQ - 1);
    size_t base = (((size_t)b_ * N_HEADS + h) * SEQ + s_) * D_HEAD;
    P[base + ni0 * 16 + lr] = (bf16t)((c0[r] + bias0) * osc);
    P[base + ni1 * 16 + lr] = (bf16t)((c1[r] + bias1) * osc);
  }
}

// ---------------- Flash attention ----------------------------------------
// grid: (SEQ/128, H, B)  block: 256. Each wave owns 16 query rows.
// 64-key blocks: 16 WMMAs per stage per wave.
__global__ __launch_bounds__(256) void attn_fwd(
    const bf16t* __restrict__ qbuf, const bf16t* __restrict__ kbuf,
    const bf16t* __restrict__ vbuf, bf16t* __restrict__ z) {
  __shared__ __align__(64) bf16t Kl[64][64];        // K block [key][dh]
  __shared__ __align__(64) bf16t Vt[64][64];        // V transposed [dh][key]
  __shared__ __align__(64) bf16t Pl[8][16][64];     // per-wave P tiles

  const int t    = threadIdx.x;
  const int wv   = t >> 5;
  const int lane = t & 31;
  const int half = lane >> 4;
  const int lr   = lane & 15;
  const int h    = blockIdx.y;
  const int b    = blockIdx.z;
  const int q0   = blockIdx.x * 128;

  const size_t headbase = ((size_t)b * N_HEADS + h) * SEQ * D_HEAD;

  const bf16t* qptr = qbuf + headbase + (size_t)(q0 + wv * 16 + lr) * D_HEAD;
  v16bf qf0 = load_afrag(qptr, half);
  v16bf qf1 = load_afrag(qptr + 32, half);

  v8f o0 = zero_v8f(), o1 = zero_v8f(), o2 = zero_v8f(), o3 = zero_v8f();
  float rm[8], rs[8];
#pragma unroll
  for (int r = 0; r < 8; ++r) { rm[r] = -1e30f; rs[r] = 0.0f; }

  const int myq  = q0 + wv * 16;
  const int kmax = q0 + 128;                        // causal block bound

  for (int kbk = 0; kbk < kmax; kbk += 64) {
    __syncthreads();
    { // stage K natural (async): 64 keys x 64 dh, 32B per thread
      int row = t >> 2, c = (t & 3) * 16;
      const bf16t* kp = &kbuf[headbase + (size_t)(kbk + row) * D_HEAD + c];
      stage16(&Kl[row][c], kp);
      stage16(&Kl[row][c + 8], kp + 8);
      if (kbk + 64 < kmax) __builtin_prefetch(kp + 64 * D_HEAD, 0, 1);
    }
    { // stage V transposed, packed b32: key rows 2p,2p+1
      int p = t >> 3, ch = (t & 7) * 8;
      const bf16t* vp = &vbuf[headbase + (size_t)(kbk + 2 * p) * D_HEAD + ch];
      v8bf r0 = *(const v8bf*)vp;
      v8bf r1 = *(const v8bf*)(vp + D_HEAD);
#pragma unroll
      for (int j = 0; j < 8; ++j)
        *(unsigned*)&Vt[ch + j][2 * p] = pack2(r0[j], r1[j]);
      if (kbk + 64 < kmax) __builtin_prefetch(vp + 64 * D_HEAD, 0, 1);
    }
    wait_stage();
    __syncthreads();

    // scores: S = Q K^T  (four 16-key groups x two dh halves)
    v8f s0 = zero_v8f(), s1 = zero_v8f(), s2 = zero_v8f(), s3 = zero_v8f();
    s0 = wmma_bf16(qf0, load_bfrag(&Kl[lr][0],       half), s0);
    s0 = wmma_bf16(qf1, load_bfrag(&Kl[lr][32],      half), s0);
    s1 = wmma_bf16(qf0, load_bfrag(&Kl[16 + lr][0],  half), s1);
    s1 = wmma_bf16(qf1, load_bfrag(&Kl[16 + lr][32], half), s1);
    s2 = wmma_bf16(qf0, load_bfrag(&Kl[32 + lr][0],  half), s2);
    s2 = wmma_bf16(qf1, load_bfrag(&Kl[32 + lr][32], half), s2);
    s3 = wmma_bf16(qf0, load_bfrag(&Kl[48 + lr][0],  half), s3);
    s3 = wmma_bf16(qf1, load_bfrag(&Kl[48 + lr][32], half), s3);

    // causal mask + online softmax (row = r + 8*half, key col = lane&15)
#pragma unroll
    for (int r = 0; r < 8; ++r) {
      int   qg = myq + r + 8 * half;
      float e0 = (kbk + lr      > qg) ? IGNORE_VAL : s0[r];
      float e1 = (kbk + 16 + lr > qg) ? IGNORE_VAL : s1[r];
      float e2 = (kbk + 32 + lr > qg) ? IGNORE_VAL : s2[r];
      float e3 = (kbk + 48 + lr > qg) ? IGNORE_VAL : s3[r];
      float bm = fmaxf(fmaxf(e0, e1), fmaxf(e2, e3));
#pragma unroll
      for (int off = 8; off >= 1; off >>= 1)
        bm = fmaxf(bm, __shfl_xor(bm, off, 16));
      float nm   = fmaxf(rm[r], bm);
      float corr = __expf(rm[r] - nm);
      float p0 = __expf(e0 - nm), p1 = __expf(e1 - nm);
      float p2 = __expf(e2 - nm), p3 = __expf(e3 - nm);
      float ls = (p0 + p1) + (p2 + p3);
#pragma unroll
      for (int off = 8; off >= 1; off >>= 1)
        ls += __shfl_xor(ls, off, 16);
      rs[r] = rs[r] * corr + ls;
      rm[r] = nm;
      o0[r] *= corr; o1[r] *= corr; o2[r] *= corr; o3[r] *= corr;
      int m = r + 8 * half;
      Pl[wv][m][lr]      = (bf16t)p0;      // C-layout -> A-layout via LDS
      Pl[wv][m][16 + lr] = (bf16t)p1;
      Pl[wv][m][32 + lr] = (bf16t)p2;
      Pl[wv][m][48 + lr] = (bf16t)p3;
    }
    __syncthreads();

    // z += P @ V  (64-key K dim = two 32-chunks)
    v16bf pa0 = load_afrag(&Pl[wv][lr][0],  half);
    v16bf pa1 = load_afrag(&Pl[wv][lr][32], half);
    o0 = wmma_bf16(pa0, load_bfrag(&Vt[lr][0],       half), o0);
    o0 = wmma_bf16(pa1, load_bfrag(&Vt[lr][32],      half), o0);
    o1 = wmma_bf16(pa0, load_bfrag(&Vt[16 + lr][0],  half), o1);
    o1 = wmma_bf16(pa1, load_bfrag(&Vt[16 + lr][32], half), o1);
    o2 = wmma_bf16(pa0, load_bfrag(&Vt[32 + lr][0],  half), o2);
    o2 = wmma_bf16(pa1, load_bfrag(&Vt[32 + lr][32], half), o2);
    o3 = wmma_bf16(pa0, load_bfrag(&Vt[48 + lr][0],  half), o3);
    o3 = wmma_bf16(pa1, load_bfrag(&Vt[48 + lr][32], half), o3);
  }

  // epilogue: normalize, store z as [BS, H*64] bf16
#pragma unroll
  for (int r = 0; r < 8; ++r) {
    float inv = 1.0f / rs[r];
    int   s_  = q0 + wv * 16 + r + 8 * half;
    size_t base = ((size_t)b * SEQ + s_) * D_MODEL + h * D_HEAD;
    z[base + lr]      = (bf16t)(o0[r] * inv);
    z[base + 16 + lr] = (bf16t)(o1[r] * inv);
    z[base + 32 + lr] = (bf16t)(o2[r] * inv);
    z[base + 48 + lr] = (bf16t)(o3[r] * inv);
  }
}

// ---------------- Output projection ---------------------------------------
// grid: (BS/64, D_MODEL/64)  block: 256. out = z @ O + Ob (f32 out).
__global__ __launch_bounds__(256) void out_proj(
    const bf16t* __restrict__ z, const bf16t* __restrict__ Wo,
    const float* __restrict__ Ob, float* __restrict__ out) {
  __shared__ __align__(64) bf16t Als[64][64];
  __shared__ __align__(64) bf16t Wt[64][64];

  const int t    = threadIdx.x;
  const int wv   = t >> 5;
  const int lane = t & 31;
  const int half = lane >> 4;
  const int lr   = lane & 15;
  const int m0   = blockIdx.x * 64;
  const int n0   = blockIdx.y * 64;

  const int mi  = (wv * 2) >> 2;
  const int ni0 = (wv * 2) & 3;
  const int ni1 = ni0 + 1;

  v8f c0 = zero_v8f(), c1 = zero_v8f();

  for (int k0 = 0; k0 < D_MODEL; k0 += 64) {
    __syncthreads();
    {
      int row = t >> 2, c = (t & 3) * 16;
      const bf16t* src = &z[(size_t)(m0 + row) * D_MODEL + k0 + c];
      stage16(&Als[row][c], src);
      stage16(&Als[row][c + 8], src + 8);
    }
    {
      int p = t >> 3, ch = (t & 7) * 8;
      const bf16t* wp = &Wo[(size_t)(k0 + 2 * p) * D_MODEL + n0 + ch];
      v8bf r0 = *(const v8bf*)wp;
      v8bf r1 = *(const v8bf*)(wp + D_MODEL);
#pragma unroll
      for (int j = 0; j < 8; ++j)
        *(unsigned*)&Wt[ch + j][2 * p] = pack2(r0[j], r1[j]);
    }
    wait_stage();
    __syncthreads();

    v16bf a0 = load_afrag(&Als[mi * 16 + lr][0],  half);
    v16bf a1 = load_afrag(&Als[mi * 16 + lr][32], half);
    c0 = wmma_bf16(a0, load_bfrag(&Wt[ni0 * 16 + lr][0],  half), c0);
    c0 = wmma_bf16(a1, load_bfrag(&Wt[ni0 * 16 + lr][32], half), c0);
    c1 = wmma_bf16(a0, load_bfrag(&Wt[ni1 * 16 + lr][0],  half), c1);
    c1 = wmma_bf16(a1, load_bfrag(&Wt[ni1 * 16 + lr][32], half), c1);
  }

  const float bias0 = Ob[n0 + ni0 * 16 + lr];
  const float bias1 = Ob[n0 + ni1 * 16 + lr];
#pragma unroll
  for (int r = 0; r < 8; ++r) {
    size_t row = (size_t)(m0 + mi * 16 + r + 8 * half);
    out[row * D_MODEL + n0 + ni0 * 16 + lr] = c0[r] + bias0;
    out[row * D_MODEL + n0 + ni1 * 16 + lr] = c1[r] + bias1;
  }
}

// ---------------- Host launcher -------------------------------------------

extern "C" void kernel_launch(void* const* d_in, const int* in_sizes, int n_in,
                              void* d_out, int out_size, void* d_ws, size_t ws_size,
                              hipStream_t stream) {
  const float* resid = (const float*)d_in[0];   // normalized_resid_pre
  const float* enc   = (const float*)d_in[1];   // encoder_output
  const float* Qs    = (const float*)d_in[2];
  const float* Qbs   = (const float*)d_in[3];
  const float* Ks    = (const float*)d_in[4];
  const float* Kbs   = (const float*)d_in[5];
  const float* Vs    = (const float*)d_in[6];
  const float* Vbs   = (const float*)d_in[7];
  const float* O     = (const float*)d_in[8];
  const float* Ob    = (const float*)d_in[9];
  float* out = (float*)d_out;

  char* ws = (char*)d_ws;
  size_t cur = 0;
  auto alloc = [&](size_t bytes) { char* p = ws + cur; cur += bytes; return p; };
  const size_t act_b = (size_t)BS * D_MODEL * sizeof(bf16t);                // 16 MB
  const size_t w_b   = (size_t)N_HEADS * D_MODEL * D_HEAD * sizeof(bf16t); // 2 MB
  const size_t hd_b  = (size_t)BATCH * N_HEADS * SEQ * D_HEAD * sizeof(bf16t); // 16 MB

  bf16t* enc_bf   = (bf16t*)alloc(act_b);
  bf16t* resid_bf = (bf16t*)alloc(act_b);
  bf16t* wq_bf    = (bf16t*)alloc(w_b);
  bf16t* wk_bf    = (bf16t*)alloc(w_b);
  bf16t* wv_bf    = (bf16t*)alloc(w_b);
  bf16t* wo_bf    = (bf16t*)alloc(w_b);
  bf16t* q_bf     = (bf16t*)alloc(hd_b);
  bf16t* k_bf     = (bf16t*)alloc(hd_b);
  bf16t* v_bf     = (bf16t*)alloc(hd_b);
  bf16t* z_bf     = (bf16t*)alloc(act_b);
  (void)ws_size;

  auto cvt = [&](const float* s, bf16t* d, size_t n) {
    int n4 = (int)(n / 4);
    cvt_f32_bf16_v4<<<(n4 + 255) / 256, 256, 0, stream>>>(
        (const float4*)s, (v4bf*)d, n4);
  };
  cvt(enc,   enc_bf,   (size_t)BS * D_MODEL);
  cvt(resid, resid_bf, (size_t)BS * D_MODEL);
  cvt(Qs, wq_bf, (size_t)N_HEADS * D_MODEL * D_HEAD);
  cvt(Ks, wk_bf, (size_t)N_HEADS * D_MODEL * D_HEAD);
  cvt(Vs, wv_bf, (size_t)N_HEADS * D_MODEL * D_HEAD);
  cvt(O,  wo_bf, (size_t)N_HEADS * D_HEAD * D_MODEL);

  dim3 pgrid(BS / 64, N_HEADS, 3);
  proj_qkv<<<pgrid, 256, 0, stream>>>(enc_bf, resid_bf, wq_bf, wk_bf, wv_bf,
                                      Qbs, Kbs, Vbs, q_bf, k_bf, v_bf);

  dim3 agrid(SEQ / 128, N_HEADS, BATCH);
  attn_fwd<<<agrid, 256, 0, stream>>>(q_bf, k_bf, v_bf, z_bf);

  dim3 ogrid(BS / 64, D_MODEL / 64);
  out_proj<<<ogrid, 256, 0, stream>>>(z_bf, wo_bf, Ob, out);
  (void)in_sizes; (void)n_in; (void)out_size;
}
